// MHA_Model_Distil_66949950210787
// MI455X (gfx1250) — compile-verified, matrix operand order checked by reading, not run on hardware
//
#include <hip/hip_runtime.h>
#include <hip/hip_bf16.h>

// MHA: B=4, S=1024, H=16, D=64, DIM=1024, SCALE=8
#define Bv_ 4
#define Sv_ 1024
#define Hh_ 16
#define Dd_ 64
#define DIMv_ 1024
#define MT_ 4     // M-tiles per wave in the QKV GEMM (64 rows)

typedef __attribute__((ext_vector_type(16))) __bf16 v16bf;
typedef __attribute__((ext_vector_type(8)))  __bf16 v8bf;
typedef __attribute__((ext_vector_type(8)))  float  v8f;

// ---- helpers -------------------------------------------------------------

__device__ __forceinline__ __bf16 f2bf(float f) {
    unsigned u = __builtin_bit_cast(unsigned, f);
    unsigned r = u + 0x7FFFu + ((u >> 16) & 1u);   // round-to-nearest-even
    unsigned short h = (unsigned short)(r >> 16);
    return __builtin_bit_cast(__bf16, h);
}

// Load a 16x32 bf16 WMMA A/B fragment from a row-major matrix.
// ISA layout: lane L (<16) owns row L, K = {k0..k0+7} U {k0+16..k0+23};
//             lane L+16 owns row L, K = {k0+8..k0+15} U {k0+24..k0+31}.
__device__ __forceinline__ v16bf load_frag(const __bf16* __restrict__ base,
                                           int row0, int ld, int k0, int lane) {
    int l = lane & 15, hi = lane >> 4;
    const __bf16* p = base + (size_t)(row0 + l) * ld + k0 + hi * 8;
    v8bf a = *(const v8bf*)p;          // 16-byte load
    v8bf b = *(const v8bf*)(p + 16);   // 16-byte load
    return __builtin_shufflevector(a, b, 0,1,2,3,4,5,6,7,8,9,10,11,12,13,14,15);
}

__device__ __forceinline__ v8f wmma_bf16(v16bf a, v16bf b, v8f c) {
    return __builtin_amdgcn_wmma_f32_16x16x32_bf16(false, a, false, b,
                                                   (short)0, c, false, false);
}

// ---- kernel 0: f32 -> bf16 ----------------------------------------------

__global__ void cvt_f32_bf16(const float* __restrict__ src,
                             __bf16* __restrict__ dst, int n) {
    int i = blockIdx.x * blockDim.x + threadIdx.x;
    if (i < n) dst[i] = f2bf(src[i]);
}

// ---- kernel 1: fused QKV projection -------------------------------------
// C[m][n] = sum_k x[m][k] * W[n][k] (+bias). One wave owns a 64x16 output
// block (4 M-tiles): per k-step, 4 A-fragments + 3 weight fragments feed
// 12 WMMAs — weight fragments are reused across all 4 A fragments.

__global__ void __launch_bounds__(32)
qkv_proj(const __bf16* __restrict__ xh,
         const __bf16* __restrict__ wq, const __bf16* __restrict__ wk,
         const __bf16* __restrict__ wv,
         const float* __restrict__ bq, const float* __restrict__ bk,
         const float* __restrict__ bv,
         __bf16* __restrict__ qo, __bf16* __restrict__ ko,
         __bf16* __restrict__ vt) {
    const int lane = threadIdx.x;
    const int n0 = blockIdx.x * 16;          // 0..1023  (output feature)
    const int m0 = blockIdx.y * 16 * MT_;    // 0..4095  (b*S+s), 64-row block

    v8f cq[MT_], ck[MT_], cv[MT_];
#pragma unroll
    for (int t = 0; t < MT_; ++t) { cq[t] = v8f{}; ck[t] = v8f{}; cv[t] = v8f{}; }

    for (int k0 = 0; k0 < DIMv_; k0 += 32) {
        v16bf fq = load_frag(wq, n0, DIMv_, k0, lane);
        v16bf fk = load_frag(wk, n0, DIMv_, k0, lane);
        v16bf fv = load_frag(wv, n0, DIMv_, k0, lane);
#pragma unroll
        for (int t = 0; t < MT_; ++t) {
            v16bf a = load_frag(xh, m0 + t * 16, DIMv_, k0, lane);
            cq[t] = wmma_bf16(a, fq, cq[t]);
            ck[t] = wmma_bf16(a, fk, ck[t]);
            cv[t] = wmma_bf16(a, fv, cv[t]);
        }
    }

    const int l = lane & 15, hi = lane >> 4;
    const int n = n0 + l;
    const int h = n >> 6, d = n & 63;
    const float biasq = bq[n], biask = bk[n], biasv = bv[n];
#pragma unroll
    for (int t = 0; t < MT_; ++t) {
#pragma unroll
        for (int r = 0; r < 8; ++r) {
            const int m = m0 + t * 16 + r + 8 * hi;
            const int b = m >> 10, s = m & 1023;
            const size_t bh = (size_t)(b * Hh_ + h);
            qo[(bh * Sv_ + s) * Dd_ + d] = f2bf((cq[t][r] + biasq) * 0.125f);
            ko[(bh * Sv_ + s) * Dd_ + d] = f2bf(ck[t][r] + biask);
            vt[(bh * Dd_ + d) * Sv_ + s] = f2bf(cv[t][r] + biasv);  // V^T
        }
    }
}

// ---- kernel 2: attention (QK^T, masked softmax, PV) ----------------------
// One wave per (b*h, 16-query tile). Scores tile 16x1024 f32 in 64KB LDS.

__global__ void __launch_bounds__(32)
attn(const __bf16* __restrict__ qb, const __bf16* __restrict__ kb,
     const __bf16* __restrict__ vt, const long long* __restrict__ mask,
     float* __restrict__ out) {
    extern __shared__ float sS[];               // [16][1024]
    const int lane = threadIdx.x;
    const int l = lane & 15, hi = lane >> 4;
    const int bh = blockIdx.x;                  // 0..63
    const int b = bh >> 4, h = bh & 15;
    const int q0 = blockIdx.y * 16;

    const __bf16* Q = qb + (size_t)bh * Sv_ * Dd_;
    const __bf16* K = kb + (size_t)bh * Sv_ * Dd_;
    const __bf16* V = vt + (size_t)bh * Dd_ * Sv_;

    // Q fragments for the whole tile (K-dim = D = 64 -> 2 fragments)
    const v16bf aq0 = load_frag(Q, q0, Dd_, 0, lane);
    const v16bf aq1 = load_frag(Q, q0, Dd_, 32, lane);

    // scores = Q K^T, masked, into LDS
    for (int nt = 0; nt < Sv_ / 16; ++nt) {
        const int n0 = nt * 16;
        v8f acc = {};
        acc = wmma_bf16(aq0, load_frag(K, n0, Dd_, 0, lane), acc);
        acc = wmma_bf16(aq1, load_frag(K, n0, Dd_, 32, lane), acc);
        const int key = n0 + l;
        const bool pad = (mask[(size_t)b * Sv_ + key] == 0);
#pragma unroll
        for (int r = 0; r < 8; ++r) {
            const int m = r + 8 * hi;
            sS[m * Sv_ + key] = pad ? -__builtin_inff() : acc[r];
        }
    }
    __syncthreads();

    // per-row softmax stats: lanes l and l+16 split row l (512 keys each),
    // combined with cross-half shuffles so all 32 lanes stay busy.
    const float* row = sS + l * Sv_;
    const int kbase = hi * (Sv_ / 2);
    float pm = -__builtin_inff();
    for (int k = 0; k < Sv_ / 2; ++k) pm = fmaxf(pm, row[kbase + k]);
    const float mrow = fmaxf(pm, __shfl_xor(pm, 16));
    float ps = 0.f;
    for (int k = 0; k < Sv_ / 2; ++k) ps += __expf(row[kbase + k] - mrow);
    const float lrow = ps + __shfl_xor(ps, 16);
    const float inv = 1.0f / lrow;

    // ctx = softmax(scores) @ V ; P regenerated on the fly into A-fragments
    v8f a0 = {}, a1 = {}, a2 = {}, a3 = {};
    for (int k0 = 0; k0 < Sv_; k0 += 32) {
        const int base = k0 + hi * 8;
        v16bf a;
#pragma unroll
        for (int j = 0; j < 8; ++j)
            a[j] = f2bf(__expf(row[base + j] - mrow) * inv);
#pragma unroll
        for (int j = 0; j < 8; ++j)
            a[8 + j] = f2bf(__expf(row[base + 16 + j] - mrow) * inv);
        a0 = wmma_bf16(a, load_frag(V, 0,  Sv_, k0, lane), a0);
        a1 = wmma_bf16(a, load_frag(V, 16, Sv_, k0, lane), a1);
        a2 = wmma_bf16(a, load_frag(V, 32, Sv_, k0, lane), a2);
        a3 = wmma_bf16(a, load_frag(V, 48, Sv_, k0, lane), a3);
    }

    // out[(b,s,h*D+d)]
#pragma unroll
    for (int r = 0; r < 8; ++r) {
        const int m = q0 + r + 8 * hi;
        const size_t o = ((size_t)b * Sv_ + m) * DIMv_ + h * Dd_ + l;
        out[o + 0]  = a0[r];
        out[o + 16] = a1[r];
        out[o + 32] = a2[r];
        out[o + 48] = a3[r];
    }
}

// ---- launch --------------------------------------------------------------

extern "C" void kernel_launch(void* const* d_in, const int* in_sizes, int n_in,
                              void* d_out, int out_size, void* d_ws, size_t ws_size,
                              hipStream_t stream) {
    const float*     x    = (const float*)d_in[0];
    const long long* mask = (const long long*)d_in[1];
    const float*     Wq   = (const float*)d_in[2];
    const float*     bq   = (const float*)d_in[3];
    const float*     Wk   = (const float*)d_in[4];
    const float*     bk   = (const float*)d_in[5];
    const float*     Wv   = (const float*)d_in[6];
    const float*     bv   = (const float*)d_in[7];
    float* out = (float*)d_out;

    const int NX = Bv_ * Sv_ * DIMv_;   // 4,194,304
    const int NW = DIMv_ * DIMv_;       // 1,048,576

    char* ws = (char*)d_ws;
    __bf16* xh  = (__bf16*)(ws);                                   // 8 MB
    __bf16* wqh = (__bf16*)(ws + (size_t)8  * 1024 * 1024);        // 2 MB
    __bf16* wkh = (__bf16*)(ws + (size_t)10 * 1024 * 1024);        // 2 MB
    __bf16* wvh = (__bf16*)(ws + (size_t)12 * 1024 * 1024);        // 2 MB
    __bf16* qbf = (__bf16*)(ws + (size_t)14 * 1024 * 1024);        // 8 MB
    __bf16* kbf = (__bf16*)(ws + (size_t)22 * 1024 * 1024);        // 8 MB
    __bf16* vtb = (__bf16*)(ws + (size_t)30 * 1024 * 1024);        // 8 MB

    cvt_f32_bf16<<<(NX + 255) / 256, 256, 0, stream>>>(x,  xh,  NX);
    cvt_f32_bf16<<<(NW + 255) / 256, 256, 0, stream>>>(Wq, wqh, NW);
    cvt_f32_bf16<<<(NW + 255) / 256, 256, 0, stream>>>(Wk, wkh, NW);
    cvt_f32_bf16<<<(NW + 255) / 256, 256, 0, stream>>>(Wv, wvh, NW);

    // 64 n-tiles x 64 m-superblocks (64 rows each), one wave per block
    qkv_proj<<<dim3(DIMv_ / 16, (Bv_ * Sv_) / (16 * MT_)), 32, 0, stream>>>(
        xh, wqh, wkh, wvh, bq, bk, bv, qbf, kbf, vtb);

    // 64 (b,h) x 64 query tiles, one wave each, 64KB LDS score tile
    attn<<<dim3(Bv_ * Hh_, Sv_ / 16), 32, 16 * Sv_ * sizeof(float), stream>>>(
        qbf, kbf, vtb, mask, out);
}
